// VectorQuantizer_3994319585565
// MI455X (gfx1250) — matrix operand order repaired and checked by reading, not compiled, not run on hardware
//
#include <hip/hip_runtime.h>
#include <hip/hip_bf16.h>

// -------------------------------------------------------------------------
// VQ-VAE codebook lookup, fused for MI455X (gfx1250, wave32, WMMA).
//   N=131072 inputs x[N,64] (f32), K=1024 codes w[K,64] (f32).
//   argmin_k (|x-e_k|^2) == argmin_k (e2[k] - 2 x.e)   (x^2 drops out)
//   outputs: quantized_st[N,64] f32, loss (1), indices[N] (as f32)
//
// Roofline: 17.2 GFLOP vs ~68 MB HBM (~3us floor at 23.3 TB/s). The fused
// argmin (cmp+select per candidate, dual-issued) is the issue-rate floor;
// v3 targets latency hiding around it:
//   * M=32 rows per wave, 4096 waves; B tile loads feed 4 WMMAs each.
//   * Two code tiles per loop iteration -> 4 independent WMMA chains'
//     worth of loads/WMMAs issue under the previous argmin VALU tail
//     (kills the v_nop hazard bubbles and s_wait_dscnt stalls).
//   * B-pack (128 KB) + e2 (4 KB) staged in LDS once per 8-wave block:
//     the 64x-reused codebook is served at WGP-local LDS bandwidth.
// Matmul in f16 WMMA (f32 accumulate; argmin robust: code deltas O(1e-2)
// vs f16-induced dot error O(1e-6)); gather + loss in exact f32.
// -------------------------------------------------------------------------

typedef __attribute__((ext_vector_type(16))) _Float16 v16h;
typedef __attribute__((ext_vector_type(8)))  float    v8f;

#define VQ_N 131072
#define VQ_D 64
#define VQ_K 1024
#define VQ_CTILES (VQ_K / 16)        // 64 code tiles of 16 codes
#define VQ_MTILES (VQ_N / 32)        // 4096 row tiles (32 rows), one wave each
#define VQ_BPACK_BYTES (VQ_K * VQ_D * 2)          // 128 KB f16 swizzled B
#define VQ_LDS_BYTES (VQ_BPACK_BYTES + VQ_K * 4)  // + e2: 132 KB

// ---------------------------------------------------------------------
// Prep: e2[k] = |w_k|^2 ; zero loss ; pack codebook into the WMMA
// B-matrix (32x16 f16, K=32 chunk) per-lane layout: lane L of block
// b = ctile*2 + kc holds, for column code = ctile*16 + (L&15), dims
// kc*32 + 8h + {0..7, 16..23} (h = L>=16) as f16 (contiguous in memory).
// ---------------------------------------------------------------------
__global__ __launch_bounds__(256) void vq_prep(const float* __restrict__ W,
                                               float* __restrict__ e2,
                                               _Float16* __restrict__ bpack,
                                               float* __restrict__ loss_slot) {
    int t = blockIdx.x * blockDim.x + threadIdx.x;   // 4096 threads
    if (t == 0) *loss_slot = 0.0f;

    if (t < VQ_K) {
        const float4* w4 = (const float4*)(W + (size_t)t * VQ_D);
        float s = 0.0f;
#pragma unroll
        for (int i = 0; i < VQ_D / 4; ++i) {
            float4 f = w4[i];
            s += f.x * f.x + f.y * f.y + f.z * f.z + f.w * f.w;
        }
        e2[t] = s;
    }

    int blk   = t >> 5;                  // 0..127
    int lane  = t & 31;
    int ctile = blk >> 1;
    int kc    = blk & 1;
    int code  = ctile * 16 + (lane & 15);
    int kbase = kc * 32 + ((lane & 16) ? 8 : 0);
    const float* src = W + (size_t)code * VQ_D + kbase;
    _Float16* dst = bpack + (size_t)t * 16;
#pragma unroll
    for (int j = 0; j < 16; ++j) dst[j] = (_Float16)src[j < 8 ? j : j + 8];
}

// ---------------------------------------------------------------------
// Main: one wave per 32-row tile. B + e2 served from LDS (staged once
// per block). 32 iterations x (2 code tiles, 8 WMMAs), running per-lane
// (min,argmin), half-wave shfl reduction, exact-f32 gather + loss.
// ---------------------------------------------------------------------
__global__ __launch_bounds__(256) void vq_main(const float* __restrict__ X,
                                               const float* __restrict__ W,
                                               const float* __restrict__ e2,
                                               const uint4* __restrict__ bpack,
                                               float* __restrict__ outq,
                                               float* __restrict__ out_loss,
                                               float* __restrict__ out_idx) {
    extern __shared__ __align__(32) char smem[];
    uint4* sB4  = (uint4*)smem;                         // 8192 x 16 B = 128 KB
    float* sE2  = (float*)(smem + VQ_BPACK_BYTES);      // 1024 floats
    const v16h* sB = (const v16h*)smem;

    // ---- Cooperative stage: global -> LDS (shared by all 8 waves).
#pragma unroll 4
    for (int i = threadIdx.x; i < VQ_BPACK_BYTES / 16; i += 256)
        sB4[i] = bpack[i];
    for (int i = threadIdx.x; i < VQ_K; i += 256)
        sE2[i] = e2[i];
    __syncthreads();

    const int lane = threadIdx.x & 31;
    const int wave = threadIdx.x >> 5;
    const int tile = blockIdx.x * 8 + wave;          // 0..4095, exact
    const int half = lane >> 4;                      // 0: rows 0-7, 1: rows 8-15
    const int sc   = lane & 15;
    const size_t rowbase = (size_t)tile * 32;

    // ---- Build two A-sets (16x32 f16 each) per ISA layout.
    // lane<16 : elems = K {klo..klo+7, klo+16..klo+23}, klo=0; lane>=16: klo=8
    const int q = half ? 2 : 0;                      // klo/4
    v16h a0, a1, a2, a3;
#define VQ_CVT4(dst, base, f)                      \
    dst[base + 0] = (_Float16)(f).x;               \
    dst[base + 1] = (_Float16)(f).y;               \
    dst[base + 2] = (_Float16)(f).z;               \
    dst[base + 3] = (_Float16)(f).w;
    {
        const float4* xr4 = (const float4*)(X + (rowbase + sc) * VQ_D);
        float4 f0 = xr4[q + 0],  f1 = xr4[q + 1];
        float4 f2 = xr4[q + 4],  f3 = xr4[q + 5];
        float4 f4 = xr4[q + 8],  f5 = xr4[q + 9];
        float4 f6 = xr4[q + 12], f7 = xr4[q + 13];
        VQ_CVT4(a0, 0, f0) VQ_CVT4(a0, 4, f1) VQ_CVT4(a0, 8, f2) VQ_CVT4(a0, 12, f3)
        VQ_CVT4(a1, 0, f4) VQ_CVT4(a1, 4, f5) VQ_CVT4(a1, 8, f6) VQ_CVT4(a1, 12, f7)
    }
    {
        const float4* xr4 = (const float4*)(X + (rowbase + 16 + sc) * VQ_D);
        float4 f0 = xr4[q + 0],  f1 = xr4[q + 1];
        float4 f2 = xr4[q + 4],  f3 = xr4[q + 5];
        float4 f4 = xr4[q + 8],  f5 = xr4[q + 9];
        float4 f6 = xr4[q + 12], f7 = xr4[q + 13];
        VQ_CVT4(a2, 0, f0) VQ_CVT4(a2, 4, f1) VQ_CVT4(a2, 8, f2) VQ_CVT4(a2, 12, f3)
        VQ_CVT4(a3, 0, f4) VQ_CVT4(a3, 4, f5) VQ_CVT4(a3, 8, f6) VQ_CVT4(a3, 12, f7)
    }
#undef VQ_CVT4

    // ---- Sweep the codebook from LDS, two 16-code tiles per iteration.
    float minv0[8], minv1[8];
    int   mini0[8], mini1[8];
#pragma unroll
    for (int v = 0; v < 8; ++v) {
        minv0[v] = 3.4e38f; mini0[v] = 0x7fffffff;
        minv1[v] = 3.4e38f; mini1[v] = 0x7fffffff;
    }

    for (int it = 0; it < VQ_CTILES / 2; ++it) {
        const int ctA = it * 2, ctB = it * 2 + 1;
        // 4 independent 32B LDS loads
        v16h b0a = sB[(size_t)(ctA * 2 + 0) * 32 + lane];
        v16h b1a = sB[(size_t)(ctA * 2 + 1) * 32 + lane];
        v16h b0b = sB[(size_t)(ctB * 2 + 0) * 32 + lane];
        v16h b1b = sB[(size_t)(ctB * 2 + 1) * 32 + lane];
        v8f c0a = {0.f, 0.f, 0.f, 0.f, 0.f, 0.f, 0.f, 0.f};
        v8f c1a = {0.f, 0.f, 0.f, 0.f, 0.f, 0.f, 0.f, 0.f};
        v8f c0b = {0.f, 0.f, 0.f, 0.f, 0.f, 0.f, 0.f, 0.f};
        v8f c1b = {0.f, 0.f, 0.f, 0.f, 0.f, 0.f, 0.f, 0.f};
        // 4 independent accumulator chains, interleaved
        c0a = __builtin_amdgcn_wmma_f32_16x16x32_f16(false, a0, false, b0a,
                                                     (short)0, c0a, false, false);
        c1a = __builtin_amdgcn_wmma_f32_16x16x32_f16(false, a2, false, b0a,
                                                     (short)0, c1a, false, false);
        c0b = __builtin_amdgcn_wmma_f32_16x16x32_f16(false, a0, false, b0b,
                                                     (short)0, c0b, false, false);
        c1b = __builtin_amdgcn_wmma_f32_16x16x32_f16(false, a2, false, b0b,
                                                     (short)0, c1b, false, false);
        c0a = __builtin_amdgcn_wmma_f32_16x16x32_f16(false, a1, false, b1a,
                                                     (short)0, c0a, false, false);
        c1a = __builtin_amdgcn_wmma_f32_16x16x32_f16(false, a3, false, b1a,
                                                     (short)0, c1a, false, false);
        c0b = __builtin_amdgcn_wmma_f32_16x16x32_f16(false, a1, false, b1b,
                                                     (short)0, c0b, false, false);
        c1b = __builtin_amdgcn_wmma_f32_16x16x32_f16(false, a3, false, b1b,
                                                     (short)0, c1b, false, false);

        const int   colA = ctA * 16 + sc;
        const int   colB = ctB * 16 + sc;
        const float e2A  = sE2[colA];
        const float e2B  = sE2[colB];
#pragma unroll
        for (int v = 0; v < 8; ++v) {
            float d;
            d = __builtin_fmaf(-2.0f, c0a[v], e2A);   // e2 - 2*dot
            if (d < minv0[v]) { minv0[v] = d; mini0[v] = colA; }
            d = __builtin_fmaf(-2.0f, c1a[v], e2A);
            if (d < minv1[v]) { minv1[v] = d; mini1[v] = colA; }
            d = __builtin_fmaf(-2.0f, c0b[v], e2B);
            if (d < minv0[v]) { minv0[v] = d; mini0[v] = colB; }
            d = __builtin_fmaf(-2.0f, c1b[v], e2B);
            if (d < minv1[v]) { minv1[v] = d; mini1[v] = colB; }
        }
    }

    // ---- Reduce across the 16 lanes of each half (C layout: lanes 0-15
    // hold rows 0-7, lanes 16-31 rows 8-15). Tie -> lowest index to match
    // jnp.argmin's first-occurrence rule.
#pragma unroll
    for (int off = 1; off < 16; off <<= 1) {
#pragma unroll
        for (int v = 0; v < 8; ++v) {
            float ov = __shfl_xor(minv0[v], off, 32);
            int   oi = __shfl_xor(mini0[v], off, 32);
            if (ov < minv0[v] || (ov == minv0[v] && oi < mini0[v])) {
                minv0[v] = ov; mini0[v] = oi;
            }
            ov = __shfl_xor(minv1[v], off, 32);
            oi = __shfl_xor(mini1[v], off, 32);
            if (ov < minv1[v] || (ov == minv1[v] && oi < mini1[v])) {
                minv1[v] = ov; mini1[v] = oi;
            }
        }
    }

    // ---- Gather codes (exact f32), write quantized_st = x + (q - x),
    // indices, and accumulate the loss partial.
    float sq = 0.0f;
    const int rbase = half * 8;
#pragma unroll
    for (int s = 0; s < 2; ++s) {
#pragma unroll
        for (int v = 0; v < 8; ++v) {
            const size_t row = rowbase + s * 16 + rbase + v;
            const int idx = s ? mini1[v] : mini0[v];   // uniform within half
            float4 q4 = ((const float4*)(W + (size_t)idx * VQ_D))[sc];
            float4 x4 = ((const float4*)(X + row * VQ_D))[sc];
            float dx = q4.x - x4.x, dy = q4.y - x4.y,
                  dz = q4.z - x4.z, dw = q4.w - x4.w;
            float4 o;                                   // straight-through
            o.x = x4.x + dx; o.y = x4.y + dy; o.z = x4.z + dz; o.w = x4.w + dw;
            ((float4*)(outq + row * VQ_D))[sc] = o;
            sq += dx * dx + dy * dy + dz * dz + dw * dw;
            if (sc == v) out_idx[row] = (float)idx;     // one writer per half
        }
    }

    // loss = 1.25 * mean((q - x)^2); pre-scaled per wave.
#pragma unroll
    for (int off = 16; off >= 1; off >>= 1) sq += __shfl_xor(sq, off, 32);
    if (lane == 0)
        atomicAdd(out_loss, sq * (1.25f / (float)((size_t)VQ_N * VQ_D)));
}

extern "C" void kernel_launch(void* const* d_in, const int* in_sizes, int n_in,
                              void* d_out, int out_size, void* d_ws, size_t ws_size,
                              hipStream_t stream) {
    (void)in_sizes; (void)n_in; (void)out_size; (void)ws_size;
    const float* X = (const float*)d_in[0];        // [N, 64] f32
    const float* W = (const float*)d_in[1];        // [K, 64] f32

    float* outq     = (float*)d_out;                       // [N*64]
    float* out_loss = (float*)d_out + (size_t)VQ_N * VQ_D; // [1]
    float* out_idx  = out_loss + 1;                        // [N] (as f32)

    float*    e2    = (float*)d_ws;                        // 4 KB
    _Float16* bpack = (_Float16*)((char*)d_ws + 4096);     // 128 KB swizzled B

    vq_prep<<<16, 256, 0, stream>>>(W, e2, bpack, out_loss);
    vq_main<<<VQ_MTILES / 8, 256, VQ_LDS_BYTES, stream>>>(
        X, W, e2, (const uint4*)bpack, outq, out_loss, out_idx);
}
